// GraceModel_824633721720
// MI455X (gfx1250) — compile-verified, matrix operand order checked by reading, not run on hardware
//
#include <hip/hip_runtime.h>

typedef __attribute__((ext_vector_type(16))) __bf16 v16bf;
typedef __attribute__((ext_vector_type(8)))  float  v8f;

// ---------------- degree / symmetric norm ----------------
__global__ void k_init_deg(float* __restrict__ deg, int N) {
    int i = blockIdx.x * blockDim.x + threadIdx.x;
    if (i < N) deg[i] = 1.0f;              // self-loop contributes 1 to every degree
}

__global__ void k_deg_accum(const long long* __restrict__ dst, float* __restrict__ deg, int E) {
    int e = blockIdx.x * blockDim.x + threadIdx.x;
    if (e < E) unsafeAtomicAdd(&deg[(int)dst[e]], 1.0f);
}

__global__ void k_dinv(float* __restrict__ deg, int N) {
    int i = blockIdx.x * blockDim.x + threadIdx.x;
    if (i < N) deg[i] = rsqrtf(deg[i]);    // deg >= 1 always (self-loops)
}

// ---------------- bf16 WMMA GEMM: Y[M,F] = X[M,K] @ W[K,F] ----------------
// One wave per 16x16 output tile. A/B fragments built per CDNA5 ISA 16-bit layouts:
//   A (16x32, MxK): lane L -> row m = L&15; half = L>>4; VGPR v holds K pair
//       k0 = (v<4 ? 2v : 16+2(v-4)) + 8*half
//   B (32x16, KxN): lane L -> col n = L&15; same K packing per half-wave.
//   C/D (16x16 f32): acc[r] is element (m = r + 8*half, n = L&15).
__global__ void k_gemm_bf16(const float* __restrict__ X, const float* __restrict__ W,
                            float* __restrict__ Y, int M, int K, int F) {
    int wavesPerBlock = blockDim.x >> 5;
    int wave = blockIdx.x * wavesPerBlock + (threadIdx.x >> 5);
    int lane = threadIdx.x & 31;
    int tilesPerRow = F >> 4;
    int mt = wave / tilesPerRow;
    int nt = wave - mt * tilesPerRow;
    if (mt * 16 >= M) return;              // wave-uniform: EXEC stays all-ones for WMMA
    int row0 = mt << 4, col0 = nt << 4;
    int half = lane >> 4;                  // 0/1
    int mr   = lane & 15;                  // A row / B,C column index

    v8f acc = {};
    for (int kb = 0; kb < K; kb += 32) {
        v16bf a, b;
#pragma unroll
        for (int v = 0; v < 8; ++v) {
            int k0 = ((v < 4) ? (2 * v) : (16 + 2 * (v - 4))) + (half << 3);
            // A fragment: X[row0+mr][kb+k0 .. +1] (contiguous -> float2 load)
            const float2 xa = *(const float2*)(X + (size_t)(row0 + mr) * K + kb + k0);
            a[2 * v]     = (__bf16)xa.x;
            a[2 * v + 1] = (__bf16)xa.y;
            // B fragment: W[kb+k0][col0+mr], W[kb+k0+1][col0+mr]
            b[2 * v]     = (__bf16)W[(size_t)(kb + k0) * F + col0 + mr];
            b[2 * v + 1] = (__bf16)W[(size_t)(kb + k0 + 1) * F + col0 + mr];
        }
        acc = __builtin_amdgcn_wmma_f32_16x16x32_bf16(false, a, false, b,
                                                      (short)0, acc, false, false);
    }
#pragma unroll
    for (int r = 0; r < 8; ++r)
        Y[(size_t)(row0 + r + (half << 3)) * F + col0 + mr] = acc[r];
}

// ---------------- self-loop term initializes the output ----------------
// out[n][f] = xw[n][f] * dinv[n]^2   (self-loop edge; also deterministically
// overwrites the whole output buffer each call)
__global__ void k_selfloop(const float* __restrict__ xw, const float* __restrict__ dinv,
                           float* __restrict__ out, long long total4, int F, int sh) {
    long long idx = (long long)blockIdx.x * blockDim.x + threadIdx.x;
    if (idx >= total4) return;
    int c4m1 = (1 << sh) - 1;
    int n = (int)(idx >> sh);
    int c = ((int)idx & c4m1) << 2;
    float s = dinv[n]; s = s * s;
    float4 v = *(const float4*)(xw + (size_t)n * F + c);
    v.x *= s; v.y *= s; v.z *= s; v.w *= s;
    *(float4*)(out + (size_t)n * F + c) = v;
}

// ---------------- edge scatter-add (dominant traffic, L2-resident) ----------------
__global__ void k_scatter(const float* __restrict__ xw, const long long* __restrict__ src,
                          const long long* __restrict__ dst, const float* __restrict__ dinv,
                          float* __restrict__ out, long long total4, int F, int sh) {
    long long idx = (long long)blockIdx.x * blockDim.x + threadIdx.x;
    if (idx >= total4) return;
    int c4m1 = (1 << sh) - 1;
    int e = (int)(idx >> sh);
    int c = ((int)idx & c4m1) << 2;
    int s = (int)src[e];
    int d = (int)dst[e];
    float nrm = dinv[s] * dinv[d];
    float4 v = *(const float4*)(xw + (size_t)s * F + c);   // coalesced gather (L2 hit)
    float* o = out + (size_t)d * F + c;
    unsafeAtomicAdd(o + 0, v.x * nrm);                     // global_atomic_add_f32
    unsafeAtomicAdd(o + 1, v.y * nrm);
    unsafeAtomicAdd(o + 2, v.z * nrm);
    unsafeAtomicAdd(o + 3, v.w * nrm);
}

// ---------------- fused bias + ReLU (in place) ----------------
__global__ void k_bias_relu(float* __restrict__ out, const float* __restrict__ b,
                            long long total4, int F, int sh) {
    long long idx = (long long)blockIdx.x * blockDim.x + threadIdx.x;
    if (idx >= total4) return;
    int c4m1 = (1 << sh) - 1;
    int n = (int)(idx >> sh);
    int c = ((int)idx & c4m1) << 2;
    float4 v  = *(float4*)(out + (size_t)n * F + c);
    float4 bb = *(const float4*)(b + c);
    v.x = fmaxf(v.x + bb.x, 0.0f);
    v.y = fmaxf(v.y + bb.y, 0.0f);
    v.z = fmaxf(v.z + bb.z, 0.0f);
    v.w = fmaxf(v.w + bb.w, 0.0f);
    *(float4*)(out + (size_t)n * F + c) = v;
}

static inline int blocks_for(long long n, int t) { return (int)((n + t - 1) / t); }

extern "C" void kernel_launch(void* const* d_in, const int* in_sizes, int n_in,
                              void* d_out, int out_size, void* d_ws, size_t ws_size,
                              hipStream_t stream) {
    const float*     x   = (const float*)d_in[0];
    const float*     W1  = (const float*)d_in[1];
    const float*     b1  = (const float*)d_in[2];
    const float*     W2  = (const float*)d_in[3];
    const float*     b2  = (const float*)d_in[4];
    const long long* ei  = (const long long*)d_in[5];

    const int F_in  = 128;
    const int H     = in_sizes[2];           // 256
    const int F_out = in_sizes[4];           // 128
    const int N     = in_sizes[0] / F_in;    // 50000
    const int E     = in_sizes[5] / 2;       // 800000
    const long long* src = ei;
    const long long* dst = ei + E;

    // workspace: dinv | xw (N*H) | h (N*H)
    char*  ws   = (char*)d_ws;
    float* dinv = (float*)ws;
    size_t off1 = ((size_t)N * 4 + 255) & ~(size_t)255;
    float* xw   = (float*)(ws + off1);
    size_t off2 = off1 + (((size_t)N * H * 4 + 255) & ~(size_t)255);
    float* h    = (float*)(ws + off2);
    float* out  = (float*)d_out;

    // ---- normalization coefficients ----
    k_init_deg <<<blocks_for(N, 256), 256, 0, stream>>>(dinv, N);
    k_deg_accum<<<blocks_for(E, 256), 256, 0, stream>>>(dst, dinv, E);
    k_dinv     <<<blocks_for(N, 256), 256, 0, stream>>>(dinv, N);

    // ---- layer 1: h = relu(Agg(x @ W1) + b1) ----
    {
        int tiles = (N / 16) * (H / 16);
        k_gemm_bf16<<<(tiles + 7) / 8, 256, 0, stream>>>(x, W1, xw, N, F_in, H);
        int sh = __builtin_ctz(H / 4);                 // H=256 -> c4=64, sh=6
        long long n4 = (long long)N * (H / 4);
        long long e4 = (long long)E * (H / 4);
        k_selfloop <<<blocks_for(n4, 256), 256, 0, stream>>>(xw, dinv, h, n4, H, sh);
        k_scatter  <<<blocks_for(e4, 256), 256, 0, stream>>>(xw, src, dst, dinv, h, e4, H, sh);
        k_bias_relu<<<blocks_for(n4, 256), 256, 0, stream>>>(h, b1, n4, H, sh);
    }

    // ---- layer 2: z = relu(Agg(h @ W2) + b2) ----
    {
        int tiles = (N / 16) * (F_out / 16);
        k_gemm_bf16<<<(tiles + 7) / 8, 256, 0, stream>>>(h, W2, xw, N, H, F_out);
        int sh = __builtin_ctz(F_out / 4);             // F_out=128 -> c4=32, sh=5
        long long n4 = (long long)N * (F_out / 4);
        long long e4 = (long long)E * (F_out / 4);
        k_selfloop <<<blocks_for(n4, 256), 256, 0, stream>>>(xw, dinv, out, n4, F_out, sh);
        k_scatter  <<<blocks_for(e4, 256), 256, 0, stream>>>(xw, src, dst, dinv, out, e4, F_out, sh);
        k_bias_relu<<<blocks_for(n4, 256), 256, 0, stream>>>(out, b2, n4, F_out, sh);
    }
}
